// SemiCurrSinkhornKnopp_53626961658390
// MI455X (gfx1250) — matrix operand order repaired and checked by reading, not compiled, not run on hardware
//
#include <hip/hip_runtime.h>

// ---------------------------------------------------------------------------
// Semi-relaxed Sinkhorn-Knopp for MI455X (gfx1250).
//   B=16 batches, n=4096 rows, k=1024 cols (+1 dustbin), eps=0.1, rho=0.5.
//
// Strategy:
//   * Row-stabilized kernel Q'[i,j] = exp(10*(logit - rowmax)) stored in BF16:
//     128 MB -> resident in the 192 MB L2 for the entire 100-iteration loop.
//   * Dustbin column D_i = (sum exp(logit-rowmax))^10 kept exact in f32.
//   * u = Q'·b via v_wmma_f32_16x16x32_bf16 (f32 accumulation in matrix pipe).
//   * v = Q'^T·a via coalesced column reduction into per-split partials,
//     summed in fixed order (fully deterministic; no float atomics).
//   * Convergence flag in workspace; all 100 iterations are launched (graph-
//     capture safe), converged iterations early-out uniformly.
// ---------------------------------------------------------------------------

#define N_ROWS  4096
#define K_COLS  1024
#define K_FULL  1025
#define N_BATCH 16
#define N_SPLIT 8            // i-splits for the Q'^T pass
#define EPS_INV 10.0f
#define FI_EXP  0.9090909090909091f   // GAMMA/(GAMMA+EPS)
#define STOPERR_F 1e-6f
#define NITER   100

typedef float  v8f  __attribute__((ext_vector_type(8)));
typedef __bf16 v8bf __attribute__((ext_vector_type(8)));
typedef __bf16 v16bf __attribute__((ext_vector_type(16)));

// ---------------- workspace layout (bytes) ----------------
#define QP_OFF  ((size_t)0)
#define QP_SZ   ((size_t)N_BATCH * N_ROWS * K_COLS * 2)        // 128 MiB bf16
#define DV_OFF  (QP_OFF + QP_SZ)                               // dustbin col f32
#define DV_SZ   ((size_t)N_BATCH * N_ROWS * 4)
#define MR_OFF  (DV_OFF + DV_SZ)                               // row max f32
#define MR_SZ   ((size_t)N_BATCH * N_ROWS * 4)
#define AP_OFF  (MR_OFF + MR_SZ)                               // a' f32
#define AP_SZ   ((size_t)N_BATCH * N_ROWS * 4)
#define BV_OFF  (AP_OFF + AP_SZ)                               // b f32 [16][1025]
#define BV_SZ   ((size_t)((N_BATCH * K_FULL * 4 + 255) / 256) * 256)
#define VP_OFF  (BV_OFF + BV_SZ)                               // v partials
#define VP_SZ   ((size_t)N_SPLIT * N_BATCH * K_COLS * 4)
#define EP_OFF  (VP_OFF + VP_SZ)                               // err partials[16]
#define EP_SZ   ((size_t)256)
#define FL_OFF  (EP_OFF + EP_SZ)                               // converged flag

// ---------------- init ----------------
__global__ void k_init(float* __restrict__ bvec, float* __restrict__ errP,
                       int* __restrict__ flag) {
  int i = blockIdx.x * 256 + threadIdx.x;
  if (i < N_BATCH * K_FULL) bvec[i] = 1.0f / (float)K_FULL;   // b0 = 1/K
  if (i < N_BATCH) errP[i] = 0.0f;
  if (i == 0) *flag = 0;
}

// ---------------- prep: rowmax, dustbin, Q' bf16 ----------------
__global__ __launch_bounds__(256) void k_prep(const float* __restrict__ logits,
                                              __bf16* __restrict__ Qp,
                                              float* __restrict__ Dv,
                                              float* __restrict__ Mr) {
  __shared__ float red[256];
  const int t = threadIdx.x;
  const size_t row = (size_t)blockIdx.y * N_ROWS + blockIdx.x;
  const float4 x = ((const float4*)(logits + row * K_COLS))[t];

  float m = fmaxf(fmaxf(x.x, x.y), fmaxf(x.z, x.w));
  red[t] = m; __syncthreads();
  for (int s = 128; s > 0; s >>= 1) { if (t < s) red[t] = fmaxf(red[t], red[t + s]); __syncthreads(); }
  m = red[0]; __syncthreads();

  float s = expf(x.x - m) + expf(x.y - m) + expf(x.z - m) + expf(x.w - m);
  red[t] = s; __syncthreads();
  for (int st = 128; st > 0; st >>= 1) { if (t < st) red[t] += red[t + st]; __syncthreads(); }
  const float S = red[0];

  union { __bf16 q[4]; uint2 u; } pk;
  pk.q[0] = (__bf16)expf(EPS_INV * (x.x - m));
  pk.q[1] = (__bf16)expf(EPS_INV * (x.y - m));
  pk.q[2] = (__bf16)expf(EPS_INV * (x.z - m));
  pk.q[3] = (__bf16)expf(EPS_INV * (x.w - m));
  ((uint2*)(Qp + row * K_COLS))[t] = pk.u;

  if (t == 0) { Dv[row] = powf(S, 10.0f); Mr[row] = m; }
}

// ---------------- step A: u = Q'·b (+ dustbin), a' = (1/n)/u  [WMMA] --------
__global__ __launch_bounds__(256) void k_a(const __bf16* __restrict__ Qp,
                                           const float* __restrict__ Dv,
                                           const float* __restrict__ bvec,
                                           float* __restrict__ aP,
                                           const int* __restrict__ flag) {
  if (*flag) return;                                // uniform early-out
  __shared__ __align__(16) __bf16 bl[K_COLS];
  __shared__ float bdust;
  const int t = threadIdx.x;
  const int batch = blockIdx.y;
  const float* bsrc = bvec + batch * K_FULL;
  for (int j = t; j < K_COLS; j += 256) bl[j] = (__bf16)bsrc[j];
  if (t == 0) bdust = bsrc[K_COLS];
  __syncthreads();

  const int wave = t >> 5, lane = t & 31;
  const int half = lane >> 4;                       // 0: K 0-7/16-23, 1: K 8-15/24-31
  const int lr = lane & 15;                         // A: row; B: column
  const int rowBase = blockIdx.x * 128 + wave * 16;
  const __bf16* qrow = Qp + ((size_t)batch * N_ROWS + rowBase + lr) * K_COLS;

  v8f acc = {0.f, 0.f, 0.f, 0.f, 0.f, 0.f, 0.f, 0.f};
  for (int kc = 0; kc < K_COLS; kc += 32) {
    // A (16x32 bf16) per ISA layout: lanes 0-15 row M, K {0..7,16..23}; +16 lanes K {8..15,24..31}
    v8bf alo = *(const v8bf*)(qrow + kc + half * 8);
    v8bf ahi = *(const v8bf*)(qrow + kc + 16 + half * 8);
    v16bf A = __builtin_shufflevector(alo, ahi, 0,1,2,3,4,5,6,7,8,9,10,11,12,13,14,15);
    // B (32x16 bf16), all 16 columns = b chunk: lanes 0-15 K {0..7,16..23}, lanes 16-31 K {8..15,24..31}
    v8bf blo = *(const v8bf*)(bl + kc + half * 8);
    v8bf bhi = *(const v8bf*)(bl + kc + 16 + half * 8);
    v16bf Bf = __builtin_shufflevector(blo, bhi, 0,1,2,3,4,5,6,7,8,9,10,11,12,13,14,15);
    acc = __builtin_amdgcn_wmma_f32_16x16x32_bf16(false, A, false, Bf,
                                                  (short)0, acc, false, false);
  }
  // D[m,col]: acc[i] holds row (half*8 + i); columns identical -> lane lr==0 writes
  if (lr == 0) {
    const size_t dbase = (size_t)batch * N_ROWS + rowBase + half * 8;
    const float bd = bdust;
#pragma unroll
    for (int i = 0; i < 8; ++i) {
      float u = acc[i] + Dv[dbase + i] * bd;
      aP[dbase + i] = (1.0f / (float)N_ROWS) / u;
    }
  }
}

// ---------------- step C: vPart[split] = (Q'^T a') partial sums -------------
__global__ __launch_bounds__(256) void k_c(const __bf16* __restrict__ Qp,
                                           const float* __restrict__ aP,
                                           float* __restrict__ vPart,
                                           const int* __restrict__ flag) {
  if (*flag) return;
  __shared__ float al[512];
  const int t = threadIdx.x;
  const int batch = blockIdx.z, isplit = blockIdx.y;
  const int i0 = isplit * 512;
  const float* asrc = aP + (size_t)batch * N_ROWS + i0;
  al[t] = asrc[t]; al[t + 256] = asrc[t + 256];
  __syncthreads();

  const int j0 = blockIdx.x * 512 + 2 * t;          // two adjacent bf16 columns
  const __bf16* q = Qp + ((size_t)batch * N_ROWS + i0) * K_COLS + j0;
  float s0 = 0.f, s1 = 0.f;
  for (int i = 0; i < 512; ++i) {
    unsigned w = *(const unsigned*)(q + (size_t)i * K_COLS);
    if ((i & 7) == 0) __builtin_prefetch(q + (size_t)(i + 16) * K_COLS, 0, 1);
    float qlo = __uint_as_float(w << 16);           // bf16 -> f32
    float qhi = __uint_as_float(w & 0xffff0000u);
    float av = al[i];
    s0 = fmaf(qlo, av, s0);
    s1 = fmaf(qhi, av, s1);
  }
  float* vp = vPart + ((size_t)isplit * N_BATCH + batch) * K_COLS;
  vp[j0] = s0; vp[j0 + 1] = s1;
}

// ---------------- step B: b <- (Pb / v)^FI, err partial ----------------------
__global__ __launch_bounds__(256) void k_b(const float* __restrict__ Dv,
                                           const float* __restrict__ aP,
                                           const float* __restrict__ vPart,
                                           float* __restrict__ bvec,
                                           float* __restrict__ errP,
                                           const int* __restrict__ flag) {
  if (*flag) return;
  __shared__ float red[256];
  const int t = threadIdx.x, batch = blockIdx.x;
  const float* dv = Dv + (size_t)batch * N_ROWS;
  const float* ap = aP + (size_t)batch * N_ROWS;
  float s = 0.f;                                    // dustbin row of Q'^T a'
  for (int i = t; i < N_ROWS; i += 256) s = fmaf(dv[i], ap[i], s);
  red[t] = s; __syncthreads();
  for (int st = 128; st > 0; st >>= 1) { if (t < st) red[t] += red[t + st]; __syncthreads(); }
  const float dustv = red[0]; __syncthreads();

  float* b = bvec + batch * K_FULL;
  float lerr = 0.f;
  for (int j = t; j < K_FULL; j += 256) {
    float bn;
    if (j < K_COLS) {
      float vj = 0.f;
#pragma unroll
      for (int sp = 0; sp < N_SPLIT; ++sp)          // fixed order: deterministic
        vj += vPart[((size_t)sp * N_BATCH + batch) * K_COLS + j];
      bn = powf((0.5f / (float)K_COLS) / vj, FI_EXP);
    } else {
      bn = 0.5f / dustv;                            // dustbin: no power
    }
    float d = bn - b[j];
    lerr += d * d;
    b[j] = bn;
  }
  red[t] = lerr; __syncthreads();
  for (int st = 128; st > 0; st >>= 1) { if (t < st) red[t] += red[t + st]; __syncthreads(); }
  if (t == 0) errP[batch] = red[0];
}

// ---------------- convergence flag ----------------
__global__ void k_err(const float* __restrict__ errP, int* __restrict__ flag) {
  if (*flag) return;
  float s = 0.f;
  for (int i = 0; i < N_BATCH; ++i) s += errP[i];   // fixed order
  if (sqrtf(s) <= STOPERR_F) *flag = 1;
}

// ---------------- final plan: n * a' * exp(10*(x-m)) * b (full f32) ---------
__global__ __launch_bounds__(256) void k_final(const float* __restrict__ logits,
                                               const float* __restrict__ Mr,
                                               const float* __restrict__ aP,
                                               const float* __restrict__ bvec,
                                               float* __restrict__ out) {
  const int t = threadIdx.x;
  const size_t row = (size_t)blockIdx.y * N_ROWS + blockIdx.x;
  const float m = Mr[row];
  const float a = aP[row] * (float)N_ROWS;
  const float* b = bvec + blockIdx.y * K_FULL;
  const float4 x = ((const float4*)(logits + row * K_COLS))[t];
  float4 r;
  r.x = a * expf(EPS_INV * (x.x - m)) * b[4 * t + 0];
  r.y = a * expf(EPS_INV * (x.y - m)) * b[4 * t + 1];
  r.z = a * expf(EPS_INV * (x.z - m)) * b[4 * t + 2];
  r.w = a * expf(EPS_INV * (x.w - m)) * b[4 * t + 3];
  ((float4*)(out + row * K_COLS))[t] = r;
}

// ---------------- host ----------------
extern "C" void kernel_launch(void* const* d_in, const int* in_sizes, int n_in,
                              void* d_out, int out_size, void* d_ws, size_t ws_size,
                              hipStream_t stream) {
  (void)in_sizes; (void)n_in; (void)out_size; (void)ws_size;
  const float* logits = (const float*)d_in[0];
  float* out = (float*)d_out;
  char* ws = (char*)d_ws;

  __bf16* Qp  = (__bf16*)(ws + QP_OFF);
  float* Dv   = (float*)(ws + DV_OFF);
  float* Mr   = (float*)(ws + MR_OFF);
  float* aP   = (float*)(ws + AP_OFF);
  float* bv   = (float*)(ws + BV_OFF);
  float* vP   = (float*)(ws + VP_OFF);
  float* errP = (float*)(ws + EP_OFF);
  int*   flag = (int*)(ws + FL_OFF);

  k_init<<<(N_BATCH * K_FULL + 255) / 256, 256, 0, stream>>>(bv, errP, flag);
  k_prep<<<dim3(N_ROWS, N_BATCH), 256, 0, stream>>>(logits, Qp, Dv, Mr);

  for (int it = 0; it < NITER; ++it) {
    k_a<<<dim3(N_ROWS / 128, N_BATCH), 256, 0, stream>>>(Qp, Dv, bv, aP, flag);
    k_c<<<dim3(K_COLS / 512, N_SPLIT, N_BATCH), 256, 0, stream>>>(Qp, aP, vP, flag);
    k_b<<<N_BATCH, 256, 0, stream>>>(Dv, aP, vP, bv, errP, flag);
    k_err<<<1, 1, 0, stream>>>(errP, flag);
  }

  k_final<<<dim3(N_ROWS, N_BATCH), 256, 0, stream>>>(logits, Mr, aP, bv, out);
}
